// HeroesCell_6098853561011
// MI455X (gfx1250) — matrix operand order repaired and checked by reading, not compiled, not run on hardware
//
#include <hip/hip_runtime.h>
#include <math.h>

// ---------------- types ----------------
typedef __attribute__((ext_vector_type(16))) __bf16 v16bf;
typedef __attribute__((ext_vector_type(8)))  __bf16 bf16x8;
typedef __attribute__((ext_vector_type(8)))  float  v8f;

#define UDIM 256

__device__ __forceinline__ float sigmoidf_(float x) { return 1.f / (1.f + expf(-x)); }

__device__ __forceinline__ v16bf combine16(bf16x8 lo, bf16x8 hi) {
  v16bf v;
#pragma unroll
  for (int i = 0; i < 8; ++i) { v[i] = lo[i]; v[i + 8] = hi[i]; }
  return v;
}

// A fragment (16x32 bf16, MxK): lane 0-15 -> M=lane, K = 0..7 then 16..23
//                               lane16-31 -> M=lane-16, K = 8..15 then 24..31
__device__ __forceinline__ v16bf load_fragA(const __bf16* p) {
  bf16x8 lo = *(const bf16x8*)(p);
  bf16x8 hi = *(const bf16x8*)(p + 16);
  return combine16(lo, hi);
}

// B fragment (32x16 bf16, KxN), W stored K-major (N x K):
// lane 0-15 -> N=lane, K = 0..7 then 16..23 ; lane16-31 -> N=lane-16, K = 8..15 then 24..31
__device__ __forceinline__ v16bf load_fragB(const __bf16* p) {
  bf16x8 lo = *(const bf16x8*)(p);
  bf16x8 hi = *(const bf16x8*)(p + 16);
  return combine16(lo, hi);
}

#define WMMA_BF16(a, b, c) \
  __builtin_amdgcn_wmma_f32_16x16x32_bf16(false, (a), false, (b), (short)0, (c), false, false)

// ---------------- fused LSTM-cell kernel ----------------
// Each wave owns a 16x32 tile of the (B x 256) U-space and accumulates 5 WMMA
// streams: gates f,i,o,g (A = [inp | Ah], W columns j*256+u of the 1024-wide
// gate weights) and s_hat (As1@Ws1 + As2@Ws2). Epilogue applied in registers.
// Wave mapping: the 8 waves of a block share the SAME column tile (same weight
// fragments -> WGP$ hits) and cover 8 consecutive 16-row tiles (streaming A).
__global__ void cell_fused(const __bf16* __restrict__ inp, const __bf16* __restrict__ Ah,
                           const __bf16* __restrict__ WxT, const __bf16* __restrict__ WhT,
                           const __bf16* __restrict__ As1, const __bf16* __restrict__ Ws1T,
                           const __bf16* __restrict__ As2, const __bf16* __restrict__ Ws2T,
                           const float* __restrict__ bx,
                           const float* __restrict__ s_state, const float* __restrict__ extra,
                           const float* __restrict__ gnrow,
                           float* __restrict__ s_out, float* __restrict__ h_out,
                           __bf16* __restrict__ h_bf, __bf16* __restrict__ gnh_bf,
                           int B, int cell) {
  const int lane = threadIdx.x & 31;
  const int wid  = blockIdx.x * (blockDim.x >> 5) + (threadIdx.x >> 5);
  const int mtiles = B >> 4;
  const int nt = wid / mtiles;      // blocks of 8 consecutive waves share nt
  const int mt = wid - nt * mtiles;
  const int m0 = mt << 4;           // 16-row tile
  const int n0 = nt << 5;           // 32-col tile
  if (m0 >= B) return;

  v8f acc[5][2] = {};               // [f,i,o,g,shat] x [col sub-tile 0/1]

  const int ar = lane & 15;
  const int ak = (lane >> 4) << 3;  // 0 / 8
  const int bn = lane & 15;
  const int bk = (lane >> 4) << 3;  // 0 / 8

  // ---- gate streams: A = inp (K=128) then Ah (K=256) ----
#pragma unroll 1
  for (int src = 0; src < 2; ++src) {
    const __bf16* A = src ? Ah : inp;
    const __bf16* W = src ? WhT : WxT;
    const int K = src ? 256 : 128;
    const __bf16* ap = A + (size_t)(m0 + ar) * K + ak;
    const __bf16* wb0[4];
    const __bf16* wb1[4];
#pragma unroll
    for (int j = 0; j < 4; ++j) {
      wb0[j] = W + (size_t)(j * 256 + n0 + bn) * K + bk;
      wb1[j] = W + (size_t)(j * 256 + n0 + 16 + bn) * K + bk;
    }
    for (int k = 0; k < K; k += 32) {
      v16bf a0 = load_fragA(ap + k);
      v16bf bb[8];
#pragma unroll
      for (int j = 0; j < 4; ++j) {
        bb[2 * j]     = load_fragB(wb0[j] + k);
        bb[2 * j + 1] = load_fragB(wb1[j] + k);
      }
#pragma unroll
      for (int j = 0; j < 4; ++j) {
        acc[j][0] = WMMA_BF16(a0, bb[2 * j],     acc[j][0]);
        acc[j][1] = WMMA_BF16(a0, bb[2 * j + 1], acc[j][1]);
      }
    }
  }

  // ---- s_hat stream: As1@Ws1 + As2@Ws2 (K=256 each) ----
#pragma unroll 1
  for (int src = 0; src < 2; ++src) {
    const __bf16* A = src ? As2 : As1;
    const __bf16* W = src ? Ws2T : Ws1T;
    const __bf16* ap = A + (size_t)(m0 + ar) * 256 + ak;
    const __bf16* wb0 = W + (size_t)(n0 + bn) * 256 + bk;
    const __bf16* wb1 = W + (size_t)(n0 + 16 + bn) * 256 + bk;
    for (int k = 0; k < 256; k += 32) {
      v16bf a0 = load_fragA(ap + k);
      v16bf b0 = load_fragB(wb0 + k);
      v16bf b1 = load_fragB(wb1 + k);
      acc[4][0] = WMMA_BF16(a0, b0, acc[4][0]);
      acc[4][1] = WMMA_BF16(a0, b1, acc[4][1]);
    }
  }

  // ---- in-register LSTM epilogue ----
  // D layout: VGPR r: lanes0-15 -> M=m0+r, N=n0+lane ; lanes16-31 -> M=m0+8+r
  const int rbase = m0 + ((lane >> 4) << 3);
#pragma unroll
  for (int ci = 0; ci < 2; ++ci) {
    const int col = n0 + (ci << 4) + (lane & 15);
    const float bf0 = bx[col];
    const float bf1 = bx[col + 256];
    const float bf2 = bx[col + 512];
    const float bf3 = bx[col + 768];
#pragma unroll
    for (int r = 0; r < 8; ++r) {
      const int row = rbase + r;
      float f  = sigmoidf_(acc[0][ci][r] + bf0);
      float i_ = sigmoidf_(acc[1][ci][r] + bf1);
      float o  = sigmoidf_(acc[2][ci][r] + bf2);
      float gg = tanhf(acc[3][ci][r] + bf3);
      float sh = tanhf(acc[4][ci][r]);
      const size_t idx = (size_t)row * UDIM + col;
      const float sstate = s_state[idx];
      float sn, hn;
      if (cell == 0) {
        // c-cell: extra = g (B x 256 row-broadcast)
        const float gval = extra[(size_t)row * UDIM];
        sn = sh + i_ * gg + (1.f - gval) * (f * sstate);
        hn = o * tanhf(sn);
        if (gnh_bf) gnh_bf[idx] = (__bf16)(gnrow[row] * hn);
      } else {
        // v-cell: extra = H_v
        const float gn = gnrow[row];
        sn = sh + (1.f - gn) * sstate + gn * (f * sstate + i_ * gg);
        hn = (1.f - gn) * extra[idx] + gn * (o * tanhf(sn));
      }
      s_out[idx] = sn;
      h_out[idx] = hn;
      h_bf[idx] = (__bf16)hn;
    }
  }
}

// ---------------- generic WMMA GEMM (MLP layers) ----------------
// out[M,N] = act( A1[M,K1] @ W1^T[N,K1] + bias[N] ), wave = 32x32 tile.
__global__ void gemm_wmma_bf16(const __bf16* __restrict__ A1, const __bf16* __restrict__ W1, int K1,
                               const float* __restrict__ bias,
                               float* __restrict__ outF, __bf16* __restrict__ outB,
                               int M, int N, int act) {
  const int lane = threadIdx.x & 31;
  const int wid  = blockIdx.x * (blockDim.x >> 5) + (threadIdx.x >> 5);
  const int mtiles = M >> 5;
  const int nt = wid / mtiles;      // consecutive waves share nt -> weight reuse
  const int mt = wid - nt * mtiles;
  const int m0 = mt << 5;
  const int n0 = nt << 5;
  if (m0 >= M || n0 >= N) return;

  v8f c00 = {}; v8f c01 = {}; v8f c10 = {}; v8f c11 = {};

  const int ar = lane & 15;
  const int ak = (lane >> 4) << 3;
  const int bn = lane & 15;
  const int bk = (lane >> 4) << 3;

  const __bf16* a0p = A1 + (size_t)(m0 + ar) * K1 + ak;
  const __bf16* a1p = A1 + (size_t)(m0 + 16 + ar) * K1 + ak;
  const __bf16* b0p = W1 + (size_t)(n0 + bn) * K1 + bk;
  const __bf16* b1p = W1 + (size_t)(n0 + 16 + bn) * K1 + bk;
  for (int k = 0; k < K1; k += 32) {
    v16bf a0 = load_fragA(a0p + k);
    v16bf a1 = load_fragA(a1p + k);
    v16bf b0 = load_fragB(b0p + k);
    v16bf b1 = load_fragB(b1p + k);
    c00 = WMMA_BF16(a0, b0, c00);
    c01 = WMMA_BF16(a0, b1, c01);
    c10 = WMMA_BF16(a1, b0, c10);
    c11 = WMMA_BF16(a1, b1, c11);
  }

  const int col0  = n0 + (lane & 15);
  const int col1  = col0 + 16;
  const int rbase = m0 + ((lane >> 4) << 3);
  const float bias0 = bias ? bias[col0] : 0.f;
  const float bias1 = bias ? bias[col1] : 0.f;

#pragma unroll
  for (int r = 0; r < 8; ++r) {
    const int rowA = rbase + r;
    const int rowB = rbase + 16 + r;
    float v00 = c00[r] + bias0, v01 = c01[r] + bias1;
    float v10 = c10[r] + bias0, v11 = c11[r] + bias1;
    if (act == 1) {  // leaky(0.3)
      v00 = v00 >= 0.f ? v00 : 0.3f * v00;
      v01 = v01 >= 0.f ? v01 : 0.3f * v01;
      v10 = v10 >= 0.f ? v10 : 0.3f * v10;
      v11 = v11 >= 0.f ? v11 : 0.3f * v11;
    }
    if (outF) {
      outF[(size_t)rowA * N + col0] = v00;
      outF[(size_t)rowA * N + col1] = v01;
      outF[(size_t)rowB * N + col0] = v10;
      outF[(size_t)rowB * N + col1] = v11;
    }
    if (outB) {
      outB[(size_t)rowA * N + col0] = (__bf16)v00;
      outB[(size_t)rowA * N + col1] = (__bf16)v01;
      outB[(size_t)rowB * N + col0] = (__bf16)v10;
      outB[(size_t)rowB * N + col1] = (__bf16)v11;
    }
  }
}

// ---------------- prep: weight cast + transpose (K x N f32 -> N x K bf16) ----
__global__ void cast_transpose(const float* __restrict__ src, __bf16* __restrict__ dst, int K, int N) {
  int idx = blockIdx.x * blockDim.x + threadIdx.x;
  if (idx >= K * N) return;
  int k = idx / N;
  int n = idx - k * N;
  dst[(size_t)n * K + k] = (__bf16)src[idx];
}

// ---------------- prep: activations ----------------
__global__ void prep_act(const float* __restrict__ x, const float* __restrict__ Hc,
                         const float* __restrict__ Hv, const float* __restrict__ g,
                         __bf16* __restrict__ inp_bf, __bf16* __restrict__ Hc_bf,
                         __bf16* __restrict__ Hv_bf, __bf16* __restrict__ Ascc,
                         __bf16* __restrict__ Ascv, float* __restrict__ gnrow,
                         float* __restrict__ gn_out, int B) {
  int i = blockIdx.x * blockDim.x + threadIdx.x;
  if (i >= B * UDIM) return;
  int row = i >> 8;
  int col = i & 255;
  float gval = g[(size_t)row * UDIM];
  float hc = Hc[i], hv = Hv[i];
  Hc_bf[i] = (__bf16)hc;
  Hv_bf[i] = (__bf16)hv;
  Ascc[i] = (__bf16)((1.f - gval) * hc);
  Ascv[i] = (__bf16)(gval * hv);
  float click = x[(size_t)row * 129 + 128];
  float gn = (click >= 0.5f) ? 1.f : 0.f;
  gn_out[i] = gn;
  if (col < 128) inp_bf[(size_t)row * 128 + col] = (__bf16)x[(size_t)row * 129 + col];
  if (col == 0) gnrow[row] = gn;
}

// ---------------- final heads (C = 1) ----------------
__global__ void head_c(const float* __restrict__ t, const float* __restrict__ Wf,
                       const float* __restrict__ bf, float* __restrict__ out, int B) {
  int row = blockIdx.x * blockDim.x + threadIdx.x;
  if (row >= B) return;
  float acc = bf[0];
  const float* tr = t + (size_t)row * 64;
#pragma unroll
  for (int j = 0; j < 64; ++j) acc += tr[j] * Wf[j];
  out[row] = sigmoidf_(acc);
}

__global__ void head_v(const float* __restrict__ t, const float* __restrict__ Wf,
                       const float* __restrict__ bf, const float* __restrict__ hcp,
                       float* __restrict__ out, int B) {
  int row = blockIdx.x * blockDim.x + threadIdx.x;
  if (row >= B) return;
  float acc = bf[0];
  const float* tr = t + (size_t)row * 64;
#pragma unroll
  for (int j = 0; j < 64; ++j) acc += tr[j] * Wf[j];
  out[row] = sigmoidf_(acc) * hcp[row];
}

// ---------------- launch ----------------
extern "C" void kernel_launch(void* const* d_in, const int* in_sizes, int n_in,
                              void* d_out, int out_size, void* d_ws, size_t ws_size,
                              hipStream_t stream) {
  const int B = in_sizes[1] / UDIM;   // H_c is (B, 256)

  const float* x    = (const float*)d_in[0];
  const float* H_c  = (const float*)d_in[1];
  const float* H_v  = (const float*)d_in[2];
  const float* s_c  = (const float*)d_in[3];
  const float* s_v  = (const float*)d_in[4];
  const float* g    = (const float*)d_in[5];
  const float* Wx_c = (const float*)d_in[6];
  const float* bx_c = (const float*)d_in[7];
  const float* Wh_c = (const float*)d_in[8];
  const float* Wx_v = (const float*)d_in[9];
  const float* bx_v = (const float*)d_in[10];
  const float* Wh_v = (const float*)d_in[11];
  const float* W_scc = (const float*)d_in[12];
  const float* W_scv = (const float*)d_in[13];
  const float* W_svv = (const float*)d_in[14];
  const float* W_svc = (const float*)d_in[15];
  const float* Wpc0 = (const float*)d_in[16];
  const float* bpc0 = (const float*)d_in[17];
  const float* Wpc1 = (const float*)d_in[18];
  const float* bpc1 = (const float*)d_in[19];
  const float* Wfcc = (const float*)d_in[20];
  const float* bfcc = (const float*)d_in[21];
  const float* Wpv0 = (const float*)d_in[22];
  const float* bpv0 = (const float*)d_in[23];
  const float* Wpv1 = (const float*)d_in[24];
  const float* bpv1 = (const float*)d_in[25];
  const float* Wfcv = (const float*)d_in[26];
  const float* bfcv = (const float*)d_in[27];

  // output layout (floats): H_c_p | H_v_p | H_c_n | H_v_n | s_c_n | s_v_n | g_n
  float* out = (float*)d_out;
  float* o_hcp = out;
  float* o_hvp = out + (size_t)B;
  float* o_hcn = out + (size_t)2 * B;
  float* o_hvn = o_hcn + (size_t)B * UDIM;
  float* o_scn = o_hvn + (size_t)B * UDIM;
  float* o_svn = o_scn + (size_t)B * UDIM;
  float* o_gn  = o_svn + (size_t)B * UDIM;

  // workspace carve-out (256B aligned)
  char* ws = (char*)d_ws;
  size_t off = 0;
  auto carve = [&](size_t bytes) -> char* {
    char* p = ws + off;
    off = (off + bytes + 255) & ~(size_t)255;
    return p;
  };

  __bf16* WxcT  = (__bf16*)carve((size_t)128 * 1024 * 2);
  __bf16* WhcT  = (__bf16*)carve((size_t)256 * 1024 * 2);
  __bf16* WxvT  = (__bf16*)carve((size_t)128 * 1024 * 2);
  __bf16* WhvT  = (__bf16*)carve((size_t)256 * 1024 * 2);
  __bf16* WsccT = (__bf16*)carve((size_t)256 * 256 * 2);
  __bf16* WscvT = (__bf16*)carve((size_t)256 * 256 * 2);
  __bf16* WsvvT = (__bf16*)carve((size_t)256 * 256 * 2);
  __bf16* WsvcT = (__bf16*)carve((size_t)256 * 256 * 2);
  __bf16* Wpc0T = (__bf16*)carve((size_t)256 * 128 * 2);
  __bf16* Wpc1T = (__bf16*)carve((size_t)128 * 64 * 2);
  __bf16* Wpv0T = (__bf16*)carve((size_t)256 * 128 * 2);
  __bf16* Wpv1T = (__bf16*)carve((size_t)128 * 64 * 2);

  __bf16* inp_bf  = (__bf16*)carve((size_t)B * 128 * 2);
  __bf16* Hc_bf   = (__bf16*)carve((size_t)B * UDIM * 2);
  __bf16* Hv_bf   = (__bf16*)carve((size_t)B * UDIM * 2);
  __bf16* Ascc    = (__bf16*)carve((size_t)B * UDIM * 2);
  __bf16* Ascv    = (__bf16*)carve((size_t)B * UDIM * 2);
  __bf16* Hcn_bf  = (__bf16*)carve((size_t)B * UDIM * 2);
  __bf16* gnHcn_bf= (__bf16*)carve((size_t)B * UDIM * 2);
  __bf16* Hvn_bf  = (__bf16*)carve((size_t)B * UDIM * 2);
  __bf16* tc0_bf  = (__bf16*)carve((size_t)B * 128 * 2);
  __bf16* tv0_bf  = (__bf16*)carve((size_t)B * 128 * 2);
  float*  gnrow   = (float*)carve((size_t)B * 4);
  float*  tc1     = (float*)carve((size_t)B * 64 * 4);
  float*  tv1     = (float*)carve((size_t)B * 64 * 4);
  (void)ws_size;

  // ---- weight prep: cast + transpose to (N x K) bf16 ----
  auto castT = [&](const float* src, __bf16* dst, int K, int N) {
    int n = K * N;
    cast_transpose<<<(n + 255) / 256, 256, 0, stream>>>(src, dst, K, N);
  };
  castT(Wx_c, WxcT, 128, 1024);
  castT(Wh_c, WhcT, 256, 1024);
  castT(Wx_v, WxvT, 128, 1024);
  castT(Wh_v, WhvT, 256, 1024);
  castT(W_scc, WsccT, 256, 256);
  castT(W_scv, WscvT, 256, 256);
  castT(W_svv, WsvvT, 256, 256);
  castT(W_svc, WsvcT, 256, 256);
  castT(Wpc0, Wpc0T, 256, 128);
  castT(Wpc1, Wpc1T, 128, 64);
  castT(Wpv0, Wpv0T, 256, 128);
  castT(Wpv1, Wpv1T, 128, 64);

  // ---- activation prep ----
  {
    int n = B * UDIM;
    prep_act<<<(n + 255) / 256, 256, 0, stream>>>(x, H_c, H_v, g, inp_bf, Hc_bf, Hv_bf,
                                                  Ascc, Ascv, gnrow, o_gn, B);
  }

  // ---- fused cells: waves = (B/16) * 8, 8 waves/block -> B/16 blocks ----
  const int cellBlocks = B / 16;
  // c-cell
  cell_fused<<<cellBlocks, 256, 0, stream>>>(inp_bf, Hc_bf, WxcT, WhcT,
                                             Ascc, WsccT, Ascv, WscvT,
                                             bx_c, s_c, g, gnrow,
                                             o_scn, o_hcn, Hcn_bf, gnHcn_bf, B, 0);
  // v-cell (depends on H_c_n via gnHcn_bf; stream order enforces it)
  cell_fused<<<cellBlocks, 256, 0, stream>>>(inp_bf, Hv_bf, WxvT, WhvT,
                                             Hv_bf, WsvvT, gnHcn_bf, WsvcT,
                                             bx_v, s_v, H_v, gnrow,
                                             o_svn, o_hvn, Hvn_bf, nullptr, B, 1);

  auto gemm = [&](const __bf16* A1, const __bf16* W1, int K1, const float* bias,
                  float* oF, __bf16* oB, int N, int act) {
    int waves = (B / 32) * (N / 32);
    int blocks = (waves + 7) / 8;
    gemm_wmma_bf16<<<blocks, 256, 0, stream>>>(A1, W1, K1, bias, oF, oB, B, N, act);
  };

  // MLP c
  gemm(Hcn_bf, Wpc0T, 256, bpc0, nullptr, tc0_bf, 128, 1);
  gemm(tc0_bf, Wpc1T, 128, bpc1, tc1, nullptr, 64, 1);
  // MLP v
  gemm(Hvn_bf, Wpv0T, 256, bpv0, nullptr, tv0_bf, 128, 1);
  gemm(tv0_bf, Wpv1T, 128, bpv1, tv1, nullptr, 64, 1);

  // heads (head_v reads H_c_p -> stream order enforces dependency)
  head_c<<<(B + 255) / 256, 256, 0, stream>>>(tc1, Wfcc, bfcc, o_hcp, B);
  head_v<<<(B + 255) / 256, 256, 0, stream>>>(tv1, Wfcv, bfcv, o_hcp, o_hvp, B);
}
